// DeformableAttention_gy_7851200217272
// MI455X (gfx1250) — matrix operand order repaired and checked by reading, not compiled, not run on hardware
//
#include <hip/hip_runtime.h>
#include <hip/hip_bf16.h>

// ---------------------------------------------------------------------------
// Deformable attention, MI455X (gfx1250, wave32, WMMA bf16 path).
// ---------------------------------------------------------------------------

typedef __attribute__((ext_vector_type(16))) __bf16 v16bf;
typedef __attribute__((ext_vector_type(8)))  __bf16 v8bf;
typedef __attribute__((ext_vector_type(8)))  float  v8f;

constexpr int B_  = 8;
constexpr int N_  = 4096;
constexpr int C_  = 256;
constexpr int S_  = 4;
constexpr int NH_ = 8;
constexpr int NP_ = 4;
constexpr int HD_ = 32;
constexpr int HG_ = 64;   // grid H = W = sqrt(N)

// workspace layout (bytes)
constexpr size_t VPM_OFF   = 0;                              // 4096*256 f32
constexpr size_t VPM_BYTES = (size_t)N_ * C_ * 4;            // 4 MiB
constexpr size_t SMP_OFF   = VPM_OFF + VPM_BYTES;            // sampled, 32 MiB
constexpr size_t SMP_BYTES = (size_t)B_ * N_ * C_ * 4;
constexpr size_t WQT_OFF   = SMP_OFF + SMP_BYTES;            // bf16 weights (transposed)
constexpr size_t WVT_OFF   = WQT_OFF + (size_t)C_ * C_ * 2;
constexpr size_t WOT_OFF   = WVT_OFF + (size_t)C_ * C_ * 2;
constexpr size_t WOFFT_OFF = WOT_OFF + (size_t)C_ * C_ * 2;
constexpr size_t WATTNT_OFF= WOFFT_OFF + (size_t)64 * C_ * 2;

// ---------------------------------------------------------------------------
// WMMA fragment helpers (16-bit operands, contiguous k order: lane half h holds k = 16h..16h+15)
// ---------------------------------------------------------------------------

// A 16x32 bf16 fragment from LDS. lanes 0-15: row m=lane, K = k0..k0+15;
// lanes 16-31: row m=lane-16, K = k0+16..k0+31.
__device__ __forceinline__ v16bf load_a_frag(const __bf16* sm, int ld, int lane, int k0) {
    const int m  = lane & 15;
    const int ko = ((lane >> 4) & 1) * 16;
    const __bf16* p = sm + m * ld + k0 + ko;
    v8bf lo = *(const v8bf*)(p);
    v8bf hi = *(const v8bf*)(p + 8);
    return __builtin_shufflevector(lo, hi, 0,1,2,3,4,5,6,7,8,9,10,11,12,13,14,15);
}

// B 32x16 bf16 fragment from a transposed (N x K, K=256) bf16 weight in global.
// lanes 0-15: col n=lane, K = k0..k0+15 ; lanes 16-31: col n=lane-16, K = k0+16..k0+31.
__device__ __forceinline__ v16bf load_b_frag(const __bf16* __restrict__ wt, int col, int k0, int lane) {
    const int kb = ((lane >> 4) & 1) * 16;
    return *(const v16bf*)(wt + (size_t)col * C_ + k0 + kb);
}

// ---------------------------------------------------------------------------
// Kernel 0: transpose + convert all weights to bf16 (B-operand friendly layout)
// ---------------------------------------------------------------------------
__global__ __launch_bounds__(256) void prep_weights_kernel(
    const float* __restrict__ Wq, const float* __restrict__ Wv,
    const float* __restrict__ Wo, const float* __restrict__ Woff,
    const float* __restrict__ Wattn,
    __bf16* __restrict__ WqT, __bf16* __restrict__ WvT, __bf16* __restrict__ WoT,
    __bf16* __restrict__ WoffT, __bf16* __restrict__ WattnT) {
    const int gid = blockIdx.x * 256 + threadIdx.x;   // 221184 total, exact grid
    if (gid < 65536) {
        int idx = gid, n = idx >> 8, k = idx & 255;
        WqT[idx] = (__bf16)Wq[k * 256 + n];
    } else if (gid < 131072) {
        int idx = gid - 65536, n = idx >> 8, k = idx & 255;
        WvT[idx] = (__bf16)Wv[k * 256 + n];
    } else if (gid < 196608) {
        int idx = gid - 131072, n = idx >> 8, k = idx & 255;
        WoT[idx] = (__bf16)Wo[k * 256 + n];
    } else if (gid < 212992) {
        int idx = gid - 196608, n = idx >> 8, k = idx & 255;
        WoffT[idx] = (__bf16)Woff[k * 64 + n];
    } else if (gid < 221184) {
        int idx = gid - 212992, n = idx >> 8, k = idx & 255;
        WattnT[idx] = (__bf16)Wattn[k * 32 + n];
    }
}

// ---------------------------------------------------------------------------
// Kernel 1: vpm = mean_S(v) @ Wv + bv   (4096 x 256)
// 16-row tile per workgroup, 8 waves, each wave -> two 16x16 output tiles.
// ---------------------------------------------------------------------------
__global__ __launch_bounds__(256) void vproj_kernel(
    const float* __restrict__ v, const __bf16* __restrict__ WvT,
    const float* __restrict__ bv, float* __restrict__ vpm) {
    __shared__ __bf16 Asm[16 * 264];
    const int tid = threadIdx.x;
    const int n0  = blockIdx.x * 16;
    for (int idx = tid; idx < 16 * 256; idx += 256) {
        int m = idx >> 8, c = idx & 255;
        size_t base = (size_t)(n0 + m) * C_ + c;
        float s = v[base] + v[(size_t)N_ * C_ + base] +
                  v[2ul * N_ * C_ + base] + v[3ul * N_ * C_ + base];
        Asm[m * 264 + c] = (__bf16)(s * 0.25f);
    }
    __syncthreads();
    const int wave = tid >> 5, lane = tid & 31;
    const int mrow = (lane & 16) ? 8 : 0;
    for (int t = 0; t < 2; ++t) {
        const int n0c = (wave * 2 + t) * 16;
        v8f acc = {};
#pragma unroll
        for (int k0 = 0; k0 < 256; k0 += 32) {
            v16bf a = load_a_frag(Asm, 264, lane, k0);
            v16bf b = load_b_frag(WvT, n0c + (lane & 15), k0, lane);
            acc = __builtin_amdgcn_wmma_f32_16x16x32_bf16(false, a, false, b,
                                                          (short)0, acc, false, false);
        }
        const int col = n0c + (lane & 15);
        const float bias = bv[col];
#pragma unroll
        for (int r = 0; r < 8; ++r)
            vpm[(size_t)(n0 + r + mrow) * C_ + col] = acc[r] + bias;
    }
}

// ---------------------------------------------------------------------------
// Kernel 2 (fused): qp = q@Wq+bq ; off = qp@Woff+boff ; w = softmax(qp@Wattn+battn)
//                   sampled = bilinear(vpm, ref+off) * w   (per 16-row tile)
// ---------------------------------------------------------------------------
__global__ __launch_bounds__(256) void qfused_kernel(
    const float* __restrict__ q,  const __bf16* __restrict__ WqT,
    const float* __restrict__ bq, const __bf16* __restrict__ WoffT,
    const float* __restrict__ boff, const __bf16* __restrict__ WattnT,
    const float* __restrict__ battn, const float* __restrict__ vpm,
    float* __restrict__ sampled) {
    __shared__ __bf16 Abf [16 * 264];
    __shared__ __bf16 QPbf[16 * 264];
    __shared__ float  OFFs[16 * 68];
    __shared__ float  ATTs[16 * 36];

    const int tid  = threadIdx.x;
    const int row0 = blockIdx.x * 16;            // flattened (b*N + n)
    for (int idx = tid; idx < 16 * 256; idx += 256) {
        int m = idx >> 8, c = idx & 255;
        Abf[m * 264 + c] = (__bf16)q[(size_t)(row0 + m) * C_ + c];
    }
    __syncthreads();

    const int wave = tid >> 5, lane = tid & 31;
    const int mrow = (lane & 16) ? 8 : 0;

    // qp GEMM -> bf16 LDS
    for (int t = 0; t < 2; ++t) {
        const int n0c = (wave * 2 + t) * 16;
        v8f acc = {};
#pragma unroll
        for (int k0 = 0; k0 < 256; k0 += 32) {
            v16bf a = load_a_frag(Abf, 264, lane, k0);
            v16bf b = load_b_frag(WqT, n0c + (lane & 15), k0, lane);
            acc = __builtin_amdgcn_wmma_f32_16x16x32_bf16(false, a, false, b,
                                                          (short)0, acc, false, false);
        }
        const int col = n0c + (lane & 15);
        const float bias = bq[col];
#pragma unroll
        for (int r = 0; r < 8; ++r)
            QPbf[(r + mrow) * 264 + col] = (__bf16)(acc[r] + bias);
    }
    __syncthreads();

    // off (4 col-tiles) + attn logits (2 col-tiles)
    if (wave < 4) {
        const int n0c = wave * 16;
        v8f acc = {};
#pragma unroll
        for (int k0 = 0; k0 < 256; k0 += 32) {
            v16bf a = load_a_frag(QPbf, 264, lane, k0);
            v16bf b = load_b_frag(WoffT, n0c + (lane & 15), k0, lane);
            acc = __builtin_amdgcn_wmma_f32_16x16x32_bf16(false, a, false, b,
                                                          (short)0, acc, false, false);
        }
        const int col = n0c + (lane & 15);
        const float bias = boff[col];
#pragma unroll
        for (int r = 0; r < 8; ++r)
            OFFs[(r + mrow) * 68 + col] = acc[r] + bias;
    } else if (wave < 6) {
        const int n0c = (wave - 4) * 16;
        v8f acc = {};
#pragma unroll
        for (int k0 = 0; k0 < 256; k0 += 32) {
            v16bf a = load_a_frag(QPbf, 264, lane, k0);
            v16bf b = load_b_frag(WattnT, n0c + (lane & 15), k0, lane);
            acc = __builtin_amdgcn_wmma_f32_16x16x32_bf16(false, a, false, b,
                                                          (short)0, acc, false, false);
        }
        const int col = n0c + (lane & 15);
        const float bias = battn[col];
#pragma unroll
        for (int r = 0; r < 8; ++r)
            ATTs[(r + mrow) * 36 + col] = acc[r] + bias;
    }
    __syncthreads();

    // sampling: 256 threads = 16 rows x 8 heads x 2 halves(16 ch each)
    const int r    = tid >> 4;
    const int h    = (tid & 15) >> 1;
    const int half = tid & 1;
    const int d0   = half * 16;
    const int row  = row0 + r;
    const int n    = row & (N_ - 1);
    const float refx = (float)(n & 63) * (1.0f / 63.0f);
    const float refy = (float)((n >> 6) & 63) * (1.0f / 63.0f);

    // softmax over NP=4
    float lg[NP_];
#pragma unroll
    for (int p = 0; p < NP_; ++p) lg[p] = ATTs[r * 36 + h * NP_ + p];
    float mx = fmaxf(fmaxf(lg[0], lg[1]), fmaxf(lg[2], lg[3]));
    float se = 0.f;
#pragma unroll
    for (int p = 0; p < NP_; ++p) { lg[p] = __expf(lg[p] - mx); se += lg[p]; }
    const float inv = 1.0f / se;

    float acc[16];
#pragma unroll
    for (int j = 0; j < 16; ++j) acc[j] = 0.f;

    auto corner = [&](int xc, int yc, float cw) {
        const bool valid = (xc >= 0) & (xc < HG_) & (yc >= 0) & (yc < HG_);
        const int xi = min(max(xc, 0), HG_ - 1);
        const int yi = min(max(yc, 0), HG_ - 1);
        const float wgt = valid ? cw : 0.f;
        const float4* src = (const float4*)(vpm + (size_t)(yi * HG_ + xi) * C_ + h * HD_ + d0);
#pragma unroll
        for (int j4 = 0; j4 < 4; ++j4) {
            float4 vv = src[j4];
            acc[j4 * 4 + 0] += wgt * vv.x;
            acc[j4 * 4 + 1] += wgt * vv.y;
            acc[j4 * 4 + 2] += wgt * vv.z;
            acc[j4 * 4 + 3] += wgt * vv.w;
        }
    };

#pragma unroll
    for (int p = 0; p < NP_; ++p) {
        const float wp = lg[p] * inv;
        const float ox = OFFs[r * 68 + (h * NP_ + p) * 2 + 0];
        const float oy = OFFs[r * 68 + (h * NP_ + p) * 2 + 1];
        const float lx = fminf(fmaxf(refx + ox, 0.f), 1.f);
        const float ly = fminf(fmaxf(refy + oy, 0.f), 1.f);
        const float x  = lx * (float)HG_ - 0.5f;
        const float y  = ly * (float)HG_ - 0.5f;
        const float x0f = floorf(x), y0f = floorf(y);
        const float tx = x - x0f, ty = y - y0f;
        const int ix = (int)x0f, iy = (int)y0f;
        corner(ix,     iy,     wp * (1.f - tx) * (1.f - ty));
        corner(ix + 1, iy,     wp * tx * (1.f - ty));
        corner(ix,     iy + 1, wp * (1.f - tx) * ty);
        corner(ix + 1, iy + 1, wp * tx * ty);
    }

    float4* dst = (float4*)(sampled + (size_t)row * C_ + h * HD_ + d0);
#pragma unroll
    for (int j4 = 0; j4 < 4; ++j4) {
        float4 o; o.x = acc[j4*4+0]; o.y = acc[j4*4+1]; o.z = acc[j4*4+2]; o.w = acc[j4*4+3];
        dst[j4] = o;
    }
}

// ---------------------------------------------------------------------------
// Kernel 3: out = sampled @ Wo + bo   (32768 x 256)
// ---------------------------------------------------------------------------
__global__ __launch_bounds__(256) void oproj_kernel(
    const float* __restrict__ sampled, const __bf16* __restrict__ WoT,
    const float* __restrict__ bo, float* __restrict__ out) {
    __shared__ __bf16 Asm[16 * 264];
    const int tid  = threadIdx.x;
    const int row0 = blockIdx.x * 16;
    for (int idx = tid; idx < 16 * 256; idx += 256) {
        int m = idx >> 8, c = idx & 255;
        Asm[m * 264 + c] = (__bf16)sampled[(size_t)(row0 + m) * C_ + c];
    }
    __syncthreads();
    const int wave = tid >> 5, lane = tid & 31;
    const int mrow = (lane & 16) ? 8 : 0;
    for (int t = 0; t < 2; ++t) {
        const int n0c = (wave * 2 + t) * 16;
        v8f acc = {};
#pragma unroll
        for (int k0 = 0; k0 < 256; k0 += 32) {
            v16bf a = load_a_frag(Asm, 264, lane, k0);
            v16bf b = load_b_frag(WoT, n0c + (lane & 15), k0, lane);
            acc = __builtin_amdgcn_wmma_f32_16x16x32_bf16(false, a, false, b,
                                                          (short)0, acc, false, false);
        }
        const int col = n0c + (lane & 15);
        const float bias = bo[col];
#pragma unroll
        for (int r = 0; r < 8; ++r)
            out[(size_t)(row0 + r + mrow) * C_ + col] = acc[r] + bias;
    }
}

// ---------------------------------------------------------------------------
extern "C" void kernel_launch(void* const* d_in, const int* in_sizes, int n_in,
                              void* d_out, int out_size, void* d_ws, size_t ws_size,
                              hipStream_t stream) {
    const float* q     = (const float*)d_in[0];
    // d_in[1] (k), d_in[5] (Wk), d_in[6] (bk) are unused by the reference.
    const float* v     = (const float*)d_in[2];
    const float* Wq    = (const float*)d_in[3];
    const float* bq    = (const float*)d_in[4];
    const float* Wv    = (const float*)d_in[7];
    const float* bv    = (const float*)d_in[8];
    const float* Wo    = (const float*)d_in[9];
    const float* bo    = (const float*)d_in[10];
    const float* Woff  = (const float*)d_in[11];
    const float* boff  = (const float*)d_in[12];
    const float* Wattn = (const float*)d_in[13];
    const float* battn = (const float*)d_in[14];
    float* out = (float*)d_out;

    char* ws = (char*)d_ws;
    float*  vpm     = (float*)(ws + VPM_OFF);
    float*  sampled = (float*)(ws + SMP_OFF);
    __bf16* WqT     = (__bf16*)(ws + WQT_OFF);
    __bf16* WvT     = (__bf16*)(ws + WVT_OFF);
    __bf16* WoT     = (__bf16*)(ws + WOT_OFF);
    __bf16* WoffT   = (__bf16*)(ws + WOFFT_OFF);
    __bf16* WattnT  = (__bf16*)(ws + WATTNT_OFF);

    prep_weights_kernel<<<864, 256, 0, stream>>>(Wq, Wv, Wo, Woff, Wattn,
                                                 WqT, WvT, WoT, WoffT, WattnT);
    vproj_kernel<<<N_ / 16, 256, 0, stream>>>(v, WvT, bv, vpm);
    qfused_kernel<<<(B_ * N_) / 16, 256, 0, stream>>>(q, WqT, bq, WoffT, boff,
                                                      WattnT, battn, vpm, sampled);
    oproj_kernel<<<(B_ * N_) / 16, 256, 0, stream>>>(sampled, WoT, bo, out);
}
